// GeometricStructureEmbedding_11957188952722
// MI455X (gfx1250) — matrix operand order repaired and checked by reading, not compiled, not run on hardware
//
#include <hip/hip_runtime.h>
#include <hip/hip_bf16.h>

typedef _Float16 v16h __attribute__((ext_vector_type(16)));
typedef _Float16 v8h  __attribute__((ext_vector_type(8)));
typedef float    v8f  __attribute__((ext_vector_type(8)));

#define NPTS     384
#define HID      256
#define NTILES   ((NPTS * NPTS) / 16)   // 9216 row tiles
#define INV_SIGD 5.0f                    // 1 / 0.2
#define FACTOR_A 3.8197186342054885f     // 180 / (15 * pi)
#define EMB_C0  (-0.07195578435503159f)  // -ln(10000)/128

// ---------------------------------------------------------------------------
// Kernel 1: 3-NN reference vectors per point.  refv[p][k][xyz] = pts[nn_k]-pts[p]
// ---------------------------------------------------------------------------
__global__ void knn_ref_kernel(const float* __restrict__ pts,
                               float* __restrict__ refv) {
  int p = blockIdx.x * blockDim.x + threadIdx.x;
  if (p >= NPTS) return;
  float px = pts[p * 3 + 0], py = pts[p * 3 + 1], pz = pts[p * 3 + 2];
  float bd[4] = {1e30f, 1e30f, 1e30f, 1e30f};
  int   bi[4] = {0, 0, 0, 0};
  for (int j = 0; j < NPTS; ++j) {
    float dx = pts[j * 3 + 0] - px;
    float dy = pts[j * 3 + 1] - py;
    float dz = pts[j * 3 + 2] - pz;
    float d2 = dx * dx + dy * dy + dz * dz;
    if (d2 < bd[3]) {                 // strict < keeps earliest index on ties
      int s = 3;
      while (s > 0 && d2 < bd[s - 1]) { bd[s] = bd[s - 1]; bi[s] = bi[s - 1]; --s; }
      bd[s] = d2; bi[s] = j;
    }
  }
  for (int k = 0; k < 3; ++k) {       // drop self (slot 0)
    int j = bi[k + 1];
    refv[(p * 3 + k) * 3 + 0] = pts[j * 3 + 0] - px;
    refv[(p * 3 + k) * 3 + 1] = pts[j * 3 + 1] - py;
    refv[(p * 3 + k) * 3 + 2] = pts[j * 3 + 2] - pz;
  }
}

// ---------------------------------------------------------------------------
// Kernel 2: cast both 256x256 weight matrices to f16 (row-major, as W[c][k])
// ---------------------------------------------------------------------------
__global__ void cvt_w_kernel(const float* __restrict__ Wd,
                             const float* __restrict__ Wa,
                             _Float16* __restrict__ Wdh,
                             _Float16* __restrict__ Wah) {
  int idx = blockIdx.x * blockDim.x + threadIdx.x;
  if (idx < HID * HID) {
    Wdh[idx] = (_Float16)Wd[idx];
    Wah[idx] = (_Float16)Wa[idx];
  }
}

// ---------------------------------------------------------------------------
// Kernel 3: main fused kernel.  One block = 16 output rows x 256 channels.
// 8 waves; each wave computes 2 column-tiles with V_WMMA_F32_16X16X32_F16.
// ---------------------------------------------------------------------------
__global__ void __launch_bounds__(256)
gse_main_kernel(const float* __restrict__ pts,
                const _Float16* __restrict__ Wdh,
                const float* __restrict__ bd,
                const _Float16* __restrict__ Wah,
                const float* __restrict__ ba,
                const float* __restrict__ refv,
                float* __restrict__ out) {
  // smem: 4 matrices [mat][row 0..15][256 halves]  (mat 0 = dist, 1..3 = angle k)
  __shared__ _Float16 smem[4 * 16 * HID];   // 32 KB
  __shared__ float    geom[64];             // [mat*16 + row] = scalar angle x

  const int tid  = threadIdx.x;
  const int r0   = blockIdx.x * 16;         // first global row of this tile

  // ---- Phase A: per-row geometry scalars (16 rows) ----
  if (tid < 16) {
    int r = r0 + tid;
    int i = r / NPTS;
    int j = r - i * NPTS;
    float pix = pts[i * 3 + 0], piy = pts[i * 3 + 1], piz = pts[i * 3 + 2];
    float ax = pts[j * 3 + 0] - pix;
    float ay = pts[j * 3 + 1] - piy;
    float az = pts[j * 3 + 2] - piz;
    geom[tid] = sqrtf(ax * ax + ay * ay + az * az) * INV_SIGD;
#pragma unroll
    for (int k = 0; k < 3; ++k) {
      float rx = refv[(i * 3 + k) * 3 + 0];
      float ry = refv[(i * 3 + k) * 3 + 1];
      float rz = refv[(i * 3 + k) * 3 + 2];
      float cx = ry * az - rz * ay;
      float cy = rz * ax - rx * az;
      float cz = rx * ay - ry * ax;
      float sn = sqrtf(cx * cx + cy * cy + cz * cz);
      float cs = rx * ax + ry * ay + rz * az;
      geom[(k + 1) * 16 + tid] = atan2f(sn, cs) * FACTOR_A;
    }
  }
  __syncthreads();

  // ---- Phase B: fill the 4 f16 embedding matrices in LDS ----
  {
    int   t   = tid & 127;                        // frequency index 0..127
    float div = __expf((float)t * EMB_C0);
    int   mr0 = (tid >> 7) * 32;                  // this thread: 32 (mat,row) slots
    unsigned int* sm32 = (unsigned int*)smem;
#pragma unroll 4
    for (int mr = mr0; mr < mr0 + 32; ++mr) {
      float ang = geom[mr] * div;
      float s = __sinf(ang);
      float c = __cosf(ang);
      unsigned int hs = (unsigned int)__builtin_bit_cast(unsigned short, (_Float16)s);
      unsigned int hc = (unsigned int)__builtin_bit_cast(unsigned short, (_Float16)c);
      sm32[mr * 128 + t] = hs | (hc << 16);       // emb[2t]=sin, emb[2t+1]=cos
    }
  }
  __syncthreads();

  // ---- Phase C: WMMA GEMMs ----
  const int wave = tid >> 5;
  const int lane = tid & 31;
  const int arow = lane & 15;                 // A-matrix row M
  const int ahi  = (lane >> 4);               // 0: K 0-15, 1: K 16-31
  const int akoff = ahi * 16;
  const int bcol  = lane & 15;                // B-matrix column within tile
  const int bkoff = ahi * 16;                 // B: lanes>=16 hold K 16..31

  v8f accD[2], accA0[2], accA1[2], accA2[2];
#pragma unroll
  for (int q = 0; q < 2; ++q)
#pragma unroll
    for (int e = 0; e < 8; ++e) {
      accD[q][e] = 0.f; accA0[q][e] = 0.f; accA1[q][e] = 0.f; accA2[q][e] = 0.f;
    }

#pragma unroll
  for (int ks = 0; ks < 8; ++ks) {
    // A fragments from LDS (shared by both column tiles of this wave)
    v16h afrag[4];
#pragma unroll
    for (int mat = 0; mat < 4; ++mat) {
      int base = mat * 4096 + arow * HID + ks * 32 + akoff;
      v8h lo = *(const v8h*)(&smem[base]);
      v8h hi = *(const v8h*)(&smem[base + 8]);
      afrag[mat] = __builtin_shufflevector(lo, hi, 0, 1, 2, 3, 4, 5, 6, 7,
                                           8, 9, 10, 11, 12, 13, 14, 15);
    }
#pragma unroll
    for (int q = 0; q < 2; ++q) {
      int c = (wave * 2 + q) * 16 + bcol;     // global output channel
      int wbase = c * HID + ks * 32 + bkoff;  // 32-byte aligned
      v16h bD = *(const v16h*)(Wdh + wbase);
      v16h bA = *(const v16h*)(Wah + wbase);
      accD[q]  = __builtin_amdgcn_wmma_f32_16x16x32_f16(false, afrag[0], false, bD,
                                                        (short)0, accD[q], false, false);
      accA0[q] = __builtin_amdgcn_wmma_f32_16x16x32_f16(false, afrag[1], false, bA,
                                                        (short)0, accA0[q], false, false);
      accA1[q] = __builtin_amdgcn_wmma_f32_16x16x32_f16(false, afrag[2], false, bA,
                                                        (short)0, accA1[q], false, false);
      accA2[q] = __builtin_amdgcn_wmma_f32_16x16x32_f16(false, afrag[3], false, bA,
                                                        (short)0, accA2[q], false, false);
    }
  }

  // ---- Epilogue: out = d_proj + bd + ba + max_k(a_proj_k) ----
  const int mbase = ahi * 8;                  // C/D layout: lanes>=16 hold M 8..15
#pragma unroll
  for (int q = 0; q < 2; ++q) {
    int c = (wave * 2 + q) * 16 + bcol;
    float bias = bd[c] + ba[c];
#pragma unroll
    for (int v = 0; v < 8; ++v) {
      float m3 = fmaxf(accA0[q][v], fmaxf(accA1[q][v], accA2[q][v]));
      out[(size_t)(r0 + mbase + v) * HID + c] = accD[q][v] + bias + m3;
    }
  }
}

// ---------------------------------------------------------------------------
extern "C" void kernel_launch(void* const* d_in, const int* in_sizes, int n_in,
                              void* d_out, int out_size, void* d_ws, size_t ws_size,
                              hipStream_t stream) {
  const float* pts = (const float*)d_in[0];   // (1,384,3)
  const float* Wd  = (const float*)d_in[1];   // (256,256)
  const float* bd  = (const float*)d_in[2];   // (256,)
  const float* Wa  = (const float*)d_in[3];   // (256,256)
  const float* ba  = (const float*)d_in[4];   // (256,)
  float* out = (float*)d_out;                 // (1,384,384,256)

  // workspace layout
  char* ws = (char*)d_ws;
  _Float16* Wdh  = (_Float16*)ws;                                  // 128 KB
  _Float16* Wah  = (_Float16*)(ws + HID * HID * sizeof(_Float16)); // 128 KB
  float*    refv = (float*)(ws + 2 * HID * HID * sizeof(_Float16)); // 384*9 f32

  knn_ref_kernel<<<3, 128, 0, stream>>>(pts, refv);
  cvt_w_kernel<<<(HID * HID + 255) / 256, 256, 0, stream>>>(Wd, Wa, Wdh, Wah);
  gse_main_kernel<<<NTILES, 256, 0, stream>>>(pts, Wdh, bd, Wah, ba, refv, out);
}
